// OPTCacheFlowAttention_43954695307551
// MI455X (gfx1250) — matrix operand order repaired and checked by reading, not compiled, not run on hardware
//
#include <hip/hip_runtime.h>

// ---------------- problem constants (match reference) ----------------
#define NH      16
#define HS      128
#define BSZ     16        // cache block size
#define NP      8
#define PL      1024
#define NDEC    64
#define MAXB    64
#define H2      (NH*HS)   // 2048
#define SCALE_F 0.08838834764831845f

// ---------------- prompt-attention tiling ----------------
#define BM 128            // q rows per workgroup (8 waves x 16)
#define BN 32             // kv tokens per tile (== WMMA K for PV)
#define KP 136            // LDS pitch for K tile  [32 tok][128 dim] bf16 (+8 pad)
#define VP 40             // LDS pitch for V^T tile [128 dim][32 tok] bf16 (+8 pad)
#define PP 40             // LDS pitch for per-wave P tile [16][32] bf16 (+8 pad)

typedef __attribute__((ext_vector_type(16))) __bf16 bf16x16;
typedef __attribute__((ext_vector_type(8)))  __bf16 bf16x8;
typedef __attribute__((ext_vector_type(8)))  float  f32x8;
typedef __attribute__((ext_vector_type(4)))  float  f32x4;

// Compose a 16-element bf16 A/B operand from two 16B LDS loads.
__device__ __forceinline__ bf16x16 lds_op16(const __bf16* p) {
    bf16x8 lo = *(const bf16x8*)p;
    bf16x8 hi = *(const bf16x8*)(p + 16);
    bf16x16 r;
#pragma unroll
    for (int i = 0; i < 8; ++i) { r[i] = lo[i]; r[8 + i] = hi[i]; }
    return r;
}

// =====================================================================
// Prompt-phase causal flash attention, bf16 WMMA (16x16x32), f32 accum.
// grid = (NP*NH, PL/BM), block = 256 (8 waves); one wave = 16 q rows.
// =====================================================================
__global__ __launch_bounds__(256)
void prompt_attn_kernel(const float* __restrict__ q,
                        const float* __restrict__ k,
                        const float* __restrict__ v,
                        float* __restrict__ out)
{
    __shared__ __align__(16) unsigned short sK[BN * KP];       // bf16 [tok][dim]
    __shared__ __align__(16) unsigned short sV[HS * VP];       // bf16 [dim][tok] (transposed)
    __shared__ __align__(16) unsigned short sP[8][16 * PP];    // per-wave P staging

    const int prompt = blockIdx.x / NH;
    const int head   = blockIdx.x % NH;
    const int q0     = blockIdx.y * BM;

    const int tid   = threadIdx.x;
    const int wave  = tid >> 5;
    const int lane  = tid & 31;
    const int lhalf = lane >> 4;       // 0: lanes 0-15, 1: lanes 16-31
    const int l16   = lane & 15;
    const int c0    = lhalf * 8;       // A-layout K offset for this lane half

    const float* qbase = q + (size_t)(prompt * PL) * H2 + head * HS;
    const float* kbase = k + (size_t)(prompt * PL) * H2 + head * HS;
    const float* vbase = v + (size_t)(prompt * PL) * H2 + head * HS;

    // ---- Q tile -> A-layout bf16 registers, SCALE folded in ----
    const int qrow = q0 + wave * 16 + l16;       // A rows: M = lane&15 (both halves)
    bf16x16 qa[4];
#pragma unroll
    for (int s = 0; s < 4; ++s) {
        const float* p0 = qbase + (size_t)qrow * H2 + s * 32 + c0;
#pragma unroll
        for (int i = 0; i < 8; ++i) {
            qa[s][i]     = (__bf16)(p0[i]      * SCALE_F);   // K = s*32 + c0 + i
            qa[s][8 + i] = (__bf16)(p0[16 + i] * SCALE_F);   // K = s*32 + c0 + 16 + i
        }
    }

    // ---- accumulators: O (16x128 f32), running max/sum per row ----
    f32x8 o[8];
#pragma unroll
    for (int d = 0; d < 8; ++d)
#pragma unroll
        for (int r = 0; r < 8; ++r) o[d][r] = 0.f;
    float mrow[8], lrow[8];
#pragma unroll
    for (int r = 0; r < 8; ++r) { mrow[r] = -1e30f; lrow[r] = 0.f; }

    const int rowbase = q0 + wave * 16 + 8 * lhalf;  // C-layout: VGPR r -> row rowbase+r
    const int myhi    = q0 + wave * 16 + 15;         // last row owned by this wave
    const int ktiles  = (q0 + BM) / BN;              // causal: tokens [0, q0+BM)

    // cooperative loader mapping: 256 threads, 16 f32 each per 32x128 tile
    const int ltk  = tid >> 3;
    const int lseg = (tid & 7) * 16;

    for (int kt = 0; kt < ktiles; ++kt) {
        const int t0 = kt * BN;
        __syncthreads();                 // previous tile fully consumed
        {
            const float* kp = kbase + (size_t)(t0 + ltk) * H2 + lseg;
            const float* vp = vbase + (size_t)(t0 + ltk) * H2 + lseg;
            float kf[16], vf[16];
#pragma unroll
            for (int i = 0; i < 16; i += 4) {
                *(f32x4*)(kf + i) = *(const f32x4*)(kp + i);
                *(f32x4*)(vf + i) = *(const f32x4*)(vp + i);
            }
            __bf16* dstK = (__bf16*)&sK[ltk * KP + lseg];
#pragma unroll
            for (int i = 0; i < 16; ++i) dstK[i] = (__bf16)kf[i];
            __bf16* dstV = (__bf16*)sV;
#pragma unroll
            for (int i = 0; i < 16; ++i) dstV[(lseg + i) * VP + ltk] = (__bf16)vf[i];
            if (kt + 1 < ktiles) {       // gfx1250 global_prefetch_b8 on next tile
                __builtin_prefetch(kp + (size_t)BN * H2, 0, 1);
                __builtin_prefetch(vp + (size_t)BN * H2, 0, 1);
            }
        }
        __syncthreads();

        if (t0 > myhi) continue;         // wave-uniform: tile fully masked

        // ---- S = (SCALE*Q) @ K^T : 16x32 via 2 N-tiles x 4 K-slices ----
        f32x8 s0, s1;
#pragma unroll
        for (int r = 0; r < 8; ++r) { s0[r] = 0.f; s1[r] = 0.f; }
        const __bf16* sKb = (const __bf16*)sK;
#pragma unroll
        for (int s = 0; s < 4; ++s) {
            // B operand: col = token, dims 8*lhalf+{0..7} and 16+8*lhalf+{0..7} per lane
            bf16x16 b0 = lds_op16(sKb + (size_t)l16        * KP + s * 32 + lhalf * 8);
            bf16x16 b1 = lds_op16(sKb + (size_t)(16 + l16) * KP + s * 32 + lhalf * 8);
            s0 = __builtin_amdgcn_wmma_f32_16x16x32_bf16(false, qa[s], false, b0,
                                                         (short)0, s0, false, false);
            s1 = __builtin_amdgcn_wmma_f32_16x16x32_bf16(false, qa[s], false, b1,
                                                         (short)0, s1, false, false);
        }

        // ---- causal mask ----
        const int tok0 = t0 + l16;
        const int tok1 = t0 + 16 + l16;
#pragma unroll
        for (int r = 0; r < 8; ++r) {
            const int rr = rowbase + r;
            if (tok0 > rr) s0[r] = -1e30f;
            if (tok1 > rr) s1[r] = -1e30f;
        }

        // ---- online softmax (row reductions across 16-lane halves) ----
#pragma unroll
        for (int r = 0; r < 8; ++r) {
            float mx = fmaxf(s0[r], s1[r]);
            mx = fmaxf(mx, __shfl_xor(mx, 1, 16));
            mx = fmaxf(mx, __shfl_xor(mx, 2, 16));
            mx = fmaxf(mx, __shfl_xor(mx, 4, 16));
            mx = fmaxf(mx, __shfl_xor(mx, 8, 16));
            const float mn    = fmaxf(mrow[r], mx);
            const float alpha = __expf(mrow[r] - mn);
            mrow[r] = mn;
            const float p0 = __expf(s0[r] - mn);
            const float p1 = __expf(s1[r] - mn);
            s0[r] = p0; s1[r] = p1;
            float sum = p0 + p1;
            sum += __shfl_xor(sum, 1, 16);
            sum += __shfl_xor(sum, 2, 16);
            sum += __shfl_xor(sum, 4, 16);
            sum += __shfl_xor(sum, 8, 16);
            lrow[r] = lrow[r] * alpha + sum;
#pragma unroll
            for (int d = 0; d < 8; ++d) o[d][r] *= alpha;
        }

        // ---- P: C-layout -> bf16 LDS -> A-layout registers ----
        __bf16* pb = (__bf16*)sP[wave];
#pragma unroll
        for (int r = 0; r < 8; ++r) {
            const int rr = r + 8 * lhalf;
            pb[rr * PP + l16]      = (__bf16)s0[r];
            pb[rr * PP + 16 + l16] = (__bf16)s1[r];
        }
        bf16x16 pa;
        {
            const __bf16* pr = pb + (size_t)l16 * PP + c0;
            bf16x8 x0 = *(const bf16x8*)pr;            // tokens c0..c0+7
            bf16x8 x1 = *(const bf16x8*)(pr + 16);     // tokens c0+16..c0+23
#pragma unroll
            for (int i = 0; i < 8; ++i) { pa[i] = x0[i]; pa[8 + i] = x1[i]; }
        }

        // ---- O += P @ V : 8 dim-tiles of 16, B from transposed V in LDS ----
        const __bf16* vb = (const __bf16*)sV;
#pragma unroll
        for (int d = 0; d < 8; ++d) {
            bf16x16 bv = lds_op16(vb + (size_t)(d * 16 + l16) * VP + lhalf * 8);
            o[d] = __builtin_amdgcn_wmma_f32_16x16x32_bf16(false, pa, false, bv,
                                                           (short)0, o[d], false, false);
        }
    }

    // ---- epilogue: normalize, store f32 ----
    float* ob = out + (size_t)(prompt * PL) * H2 + head * HS;
#pragma unroll
    for (int r = 0; r < 8; ++r) {
        const int row = rowbase + r;
        const float inv = 1.f / lrow[r];        // diag element always unmasked
        float* orow = ob + (size_t)row * H2;
#pragma unroll
        for (int d = 0; d < 8; ++d) orow[d * 16 + l16] = o[d][r] * inv;
    }
}

// =====================================================================
// Decode-phase paged attention (memory-bound, f32).
// grid = (NDEC, NH), block = 128. Virtual cache update: token ctx-1
// comes from fresh key/value rows instead of the (unwritten) cache.
// =====================================================================
__global__ __launch_bounds__(128)
void decode_attn_kernel(const float* __restrict__ q,
                        const float* __restrict__ knew,
                        const float* __restrict__ vnew,
                        const float* __restrict__ kc,
                        const float* __restrict__ vc,
                        const int*   __restrict__ bt,
                        const int*   __restrict__ ctxl,
                        float* __restrict__ out)
{
    __shared__ float qs[HS];
    __shared__ float sc[MAXB * BSZ];     // scores -> probs
    __shared__ float red[128];

    const int g   = blockIdx.x;
    const int h   = blockIdx.y;
    const int tid = threadIdx.x;
    const int ctx = ctxl[g];
    const size_t qoff = (size_t)(NP * PL + g) * H2 + (size_t)h * HS;
    const size_t hoff = (size_t)h * (HS * BSZ);

    qs[tid] = q[qoff + tid];
    __syncthreads();

    // scores: thread-per-token; lanes share a cache block -> coalesced per dim
    for (int t = tid; t < ctx; t += 128) {
        float acc = 0.f;
        if (t == ctx - 1) {
            const float* kp = knew + qoff;
#pragma unroll 8
            for (int d = 0; d < HS; ++d) acc += qs[d] * kp[d];
        } else {
            const int blk = bt[g * MAXB + (t >> 4)];
            const float* kp = kc + (size_t)blk * (NH * HS * BSZ) + hoff + (t & 15);
#pragma unroll 8
            for (int d = 0; d < HS; ++d) acc += qs[d] * kp[d * BSZ];
        }
        sc[t] = acc * SCALE_F;
    }
    __syncthreads();

    // softmax: block max, exp, block sum
    float mx = -1e30f;
    for (int t = tid; t < ctx; t += 128) mx = fmaxf(mx, sc[t]);
    red[tid] = mx;
    __syncthreads();
    for (int s = 64; s > 0; s >>= 1) {
        if (tid < s) red[tid] = fmaxf(red[tid], red[tid + s]);
        __syncthreads();
    }
    mx = red[0];
    __syncthreads();
    float lsum = 0.f;
    for (int t = tid; t < ctx; t += 128) {
        const float p = __expf(sc[t] - mx);
        sc[t] = p;
        lsum += p;
    }
    red[tid] = lsum;
    __syncthreads();
    for (int s = 64; s > 0; s >>= 1) {
        if (tid < s) red[tid] += red[tid + s];
        __syncthreads();
    }
    const float l = red[0];
    __syncthreads();

    // P@V: thread-per-dim; each lane reads a contiguous 64B run per block
    const int d = tid;
    float acc = 0.f;
    const int nctx = ctx - 1;                 // tokens sourced from the cache
    for (int tb = 0; tb < nctx; tb += BSZ) {
        const int blk = bt[g * MAXB + (tb >> 4)];
        const float* vp = vc + (size_t)blk * (NH * HS * BSZ) + hoff + (size_t)d * BSZ;
        int jn = nctx - tb; if (jn > BSZ) jn = BSZ;
        if (jn == BSZ) {
            f32x4 a = *(const f32x4*)(vp);
            f32x4 b = *(const f32x4*)(vp + 4);
            f32x4 c = *(const f32x4*)(vp + 8);
            f32x4 e = *(const f32x4*)(vp + 12);
            const float* pp = &sc[tb];
            acc += pp[0]*a[0] + pp[1]*a[1] + pp[2]*a[2]  + pp[3]*a[3];
            acc += pp[4]*b[0] + pp[5]*b[1] + pp[6]*b[2]  + pp[7]*b[3];
            acc += pp[8]*c[0] + pp[9]*c[1] + pp[10]*c[2] + pp[11]*c[3];
            acc += pp[12]*e[0]+ pp[13]*e[1]+ pp[14]*e[2] + pp[15]*e[3];
        } else {
            for (int j = 0; j < jn; ++j) acc += sc[tb + j] * vp[j];
        }
    }
    acc += sc[ctx - 1] * vnew[qoff + d];      // fresh token's V row
    out[qoff + d] = acc / l;
}

// =====================================================================
extern "C" void kernel_launch(void* const* d_in, const int* in_sizes, int n_in,
                              void* d_out, int out_size, void* d_ws, size_t ws_size,
                              hipStream_t stream) {
    (void)in_sizes; (void)n_in; (void)out_size; (void)d_ws; (void)ws_size;
    const float* q   = (const float*)d_in[0];
    const float* k   = (const float*)d_in[1];
    const float* v   = (const float*)d_in[2];
    const float* kc  = (const float*)d_in[3];
    const float* vc  = (const float*)d_in[4];
    const int*   bt  = (const int*)d_in[5];
    const int*   cl  = (const int*)d_in[6];
    float* out = (float*)d_out;

    dim3 pgrid(NP * NH, PL / BM);             // 128 x 8 workgroups
    prompt_attn_kernel<<<pgrid, 256, 0, stream>>>(q, k, v, out);

    dim3 dgrid(NDEC, NH);                     // 64 x 16 workgroups
    decode_attn_kernel<<<dgrid, 128, 0, stream>>>(q, k, v, kc, vc, bt, cl, out);
}
